// MyQuantizedModel_80616536146255
// MI455X (gfx1250) — 8-bit matrix operand order repaired and checked by reading, compiled, not run on hardware
//
#include <hip/hip_runtime.h>
#include <hip/hip_bf16.h>

typedef __attribute__((ext_vector_type(2))) int v2i;
typedef __attribute__((ext_vector_type(4))) int v4i;
typedef __attribute__((ext_vector_type(8))) int v8i;

#define S_IN   0.3f
#define S_W    0.4f
#define S_CONV (0.3f * 0.4f)
#define S_OUT  0.4f

#define NB   16
#define CI   64
#define HIN  128
#define WIN  128
#define CO   128
#define HO   126
#define WO   126

#define WTILE   64
#define LPIX    (WTILE + 2)
#define LROW    (LPIX * CI)
#define LDSA_SZ (3 * LROW)

__global__ __launch_bounds__(256) void quant_input_kernel(
    const float* __restrict__ x, const int* __restrict__ add_const,
    unsigned char* __restrict__ q2out) {
  const int n = blockIdx.x >> 7;
  const int h = blockIdx.x & 127;
  __shared__ unsigned char tile[CI * WIN];

  const float addc = (float)(*add_const);
  const float* src = x + ((size_t)n * CI * HIN * WIN) + (size_t)h * WIN;

  for (int i = threadIdx.x; i < CI * WIN; i += 256) {
    const int c = i >> 7;
    const int w = i & 127;
    float v = src[(size_t)c * HIN * WIN + w];
    float qx = fminf(fmaxf(rintf(v * (1.0f / S_IN)), 0.0f), 255.0f);
    float q2 = fminf(fmaxf(rintf((qx * S_IN + addc) * (1.0f / S_IN)), 0.0f), 255.0f);
    tile[i] = (unsigned char)q2;
  }
  __syncthreads();

  unsigned char* dst = q2out + (((size_t)n * HIN + h) * WIN) * CI;
  for (int i = threadIdx.x; i < CI * WIN; i += 256) {
    const int w = i >> 6;
    const int c = i & 63;
    dst[i] = tile[c * WIN + w];
  }
}

__global__ __launch_bounds__(256) void pack_weights_kernel(
    const float* __restrict__ wsrc, unsigned char* __restrict__ bpack) {
  const int idx = blockIdx.x * 256 + threadIdx.x;
  if (idx >= 9 * 8 * 1024) return;
  const int block  = idx >> 10;
  const int within = idx & 1023;
  const int lane = within >> 5;
  const int v    = (within >> 2) & 7;
  const int b    = within & 3;
  const int khw   = block >> 3;
  const int otile = block & 7;
  const int kh = khw / 3, kw = khw % 3;
  const int col  = lane & 15;
  const int half = lane >> 4;
  const int k = (v >> 2) * 32 + half * 16 + (v & 3) * 4 + b;
  const int o = otile * 16 + col;
  float wv = wsrc[(((size_t)o * CI + k) * 3 + kh) * 3 + kw];
  float q = fminf(fmaxf(rintf(wv * (1.0f / S_W)), 0.0f), 255.0f);
  bpack[idx] = (unsigned char)q;
}

__global__ __launch_bounds__(256) void conv_wmma_kernel(
    const unsigned char* __restrict__ q2,
    const unsigned char* __restrict__ bpack,
    const int* __restrict__ bias,
    float* __restrict__ out) {
  const int bx = blockIdx.x;
  const int wt = bx & 1;
  const int hp = (bx >> 1) % HO;
  const int n  = bx / (HO * 2);
  const int wbase = wt * WTILE;
  const int wave = threadIdx.x >> 5;
  const int lane = threadIdx.x & 31;
  const int row  = lane & 15;
  const int half = lane >> 4;

  __shared__ unsigned char ldsA[LDSA_SZ];

  for (int i = threadIdx.x; i < LDSA_SZ / 16; i += 256) {
    const int r = i / (LROW / 16);
    const int c = i % (LROW / 16);
    const size_t g = (((size_t)n * HIN + (hp + r)) * WIN + wbase) * CI +
                     (size_t)c * 16;
    *(v4i*)(&ldsA[r * LROW + c * 16]) = *(const v4i*)(q2 + g);
  }
  __syncthreads();

  v8i acc[4];
#pragma unroll
  for (int s = 0; s < 4; ++s) acc[s] = (v8i){};

  for (int kh = 0; kh < 3; ++kh) {
    for (int kw = 0; kw < 3; ++kw) {
      const unsigned char* bp =
          bpack + ((size_t)((kh * 3 + kw) * 8 + wave)) * 1024 + lane * 32;
      v4i b0 = *(const v4i*)(bp + 0);
      v4i b1 = *(const v4i*)(bp + 16);
      v8i bfrag = {b0.x, b0.y, b0.z, b0.w, b1.x, b1.y, b1.z, b1.w};

#pragma unroll
      for (int s = 0; s < 4; ++s) {
        const unsigned char* lp =
            &ldsA[kh * LROW + (s * 16 + kw) * CI + row * 64 + half * 16];
        v2i p0 = *(const v2i*)(lp + 0);
        v2i p1 = *(const v2i*)(lp + 8);
        v2i p2 = *(const v2i*)(lp + 32);
        v2i p3 = *(const v2i*)(lp + 40);
        v8i afrag = {p0.x, p0.y, p1.x, p1.y, p2.x, p2.y, p3.x, p3.y};

        acc[s] = __builtin_amdgcn_wmma_i32_16x16x64_iu8(
            false, afrag, false, bfrag, acc[s], false, false);
      }
    }
  }

  const int o = wave * 16 + (lane & 15);
  const float bterm = (float)bias[o] * (1.0f / S_CONV);
#pragma unroll
  for (int s = 0; s < 4; ++s) {
#pragma unroll
    for (int r = 0; r < 8; ++r) {
      const int wp = wbase + s * 16 + r + 8 * half;
      if (wp < WO) {
        float a    = (float)acc[s][r] + bterm;
        float q32  = rintf(a);
        float deq2 = q32 * S_CONV;
        float q    = fminf(fmaxf(rintf(deq2 * (1.0f / S_OUT)), 0.0f), 255.0f);
        out[(((size_t)n * HO + hp) * WO + wp) * CO + o] = q * S_OUT;
      }
    }
  }
}

extern "C" void kernel_launch(void* const* d_in, const int* in_sizes, int n_in,
                              void* d_out, int out_size, void* d_ws, size_t ws_size,
                              hipStream_t stream) {
  const float* x_float = (const float*)d_in[0];
  const float* weights = (const float*)d_in[1];
  const int*   bias    = (const int*)d_in[2];
  const int*   add_c   = (const int*)d_in[3];
  float*       out     = (float*)d_out;

  unsigned char* q2buf = (unsigned char*)d_ws;
  const size_t q2_bytes = (size_t)NB * HIN * WIN * CI;
  unsigned char* bpack = q2buf + q2_bytes + 1024;

  quant_input_kernel<<<NB * HIN, 256, 0, stream>>>(x_float, add_c, q2buf);
  pack_weights_kernel<<<(9 * 8 * 1024 + 255) / 256, 256, 0, stream>>>(weights, bpack);
  conv_wmma_kernel<<<NB * HO * 2, 256, 0, stream>>>(q2buf, bpack, bias, out);
}
